// InternVLA_13821204759061
// MI455X (gfx1250) — compile-verified, matrix operand order checked by reading, not run on hardware
//
#include <hip/hip_runtime.h>
#include <hip/hip_bf16.h>

// ---------------------------------------------------------------------------
// InternVLA block forward for MI455X (gfx1250), wave32, WMMA f32_16x16x32_f16.
// B=4, S1=1024, S2=64, S=1088, D=896, FF=4864, Hq=14, Hkv=2, HD=64.
// ---------------------------------------------------------------------------

#define B_     4
#define S1_    1024
#define S2_    64
#define S_     1088          // 17*64
#define NTOK   (B_ * S_)     // 4352 = 68*64
#define D_     896
#define FF_    4864
#define HQ_    14
#define HKV_   2
#define HD_    64
#define QKVN   (HQ_*HD_ + 2*HKV_*HD_)   // 1152
#define NEG_BIG (-1.0e30f)

typedef __attribute__((ext_vector_type(16))) _Float16 v16h;
typedef __attribute__((ext_vector_type(8)))  float    v8f;

// ---------------- WMMA helpers -------------------------------------------

__device__ inline v8f wmma_f16(v16h a, v16h b, v8f c) {
    // D(16x16,f32) = A(16x32,f16) x B(32x16,f16) + C
    return __builtin_amdgcn_wmma_f32_16x16x32_f16(
        false, a, false, b, (short)0, c, false, false);
}

// Load a 16x32 fragment from a row-major [16+][stride] f16 buffer.
// A-matrix: lane = row (l%16), K packed 2/VGPR, lane-half selects K octet.
// B-matrix (32x16) uses the symmetric layout with lane = column, so the same
// loader works when the buffer is stored [N][K] (K contiguous).
__device__ inline v16h ldfrag(const _Float16* base, int stride) {
    const int l  = threadIdx.x & 31;
    const int rn = l & 15;
    const int hs = (l & 16) ? 8 : 0;
    const _Float16* pr = base + rn * stride;
    v16h f;
#pragma unroll
    for (int j = 0; j < 8; ++j) {
        const int k0 = ((j >> 2) << 4) + hs + ((j & 3) << 1);
        f[2 * j]     = pr[k0];
        f[2 * j + 1] = pr[k0 + 1];
    }
    return f;
}

// half-wave (16-lane) reductions: C-fragment rows live within a lane-half.
__device__ inline float redmax16(float v) {
#pragma unroll
    for (int off = 8; off; off >>= 1) v = fmaxf(v, __shfl_xor(v, off, 16));
    return v;
}
__device__ inline float redsum16(float v) {
#pragma unroll
    for (int off = 8; off; off >>= 1) v += __shfl_xor(v, off, 16);
    return v;
}

// ---------------- RMSNorm -> f16 ------------------------------------------
// If x1 != nullptr: input is the (x0,x1) pair in reference layout.
// Else: x0 is a contiguous [NTOK, D] f32 buffer.
__global__ __launch_bounds__(256) void rmsnorm_kernel(
    const float* __restrict__ x0, const float* __restrict__ x1,
    const float* __restrict__ w /*[2,D]*/, _Float16* __restrict__ out) {
    const int t = blockIdx.x;
    const int b = t / S_;
    const int s = t - b * S_;
    const int seg = (s >= S1_) ? 1 : 0;
    const float* xr;
    if (x1) xr = seg ? x1 + (size_t)(b * S2_ + (s - S1_)) * D_
                     : x0 + (size_t)(b * S1_ + s) * D_;
    else    xr = x0 + (size_t)t * D_;
    const float* wr = w + seg * D_;

    __shared__ float red[8];
    float ss = 0.f;
    for (int d = threadIdx.x; d < D_; d += 256) { float v = xr[d]; ss += v * v; }
#pragma unroll
    for (int off = 16; off; off >>= 1) ss += __shfl_xor(ss, off, 32);
    if ((threadIdx.x & 31) == 0) red[threadIdx.x >> 5] = ss;
    __syncthreads();
    float tot = 0.f;
#pragma unroll
    for (int i = 0; i < 8; ++i) tot += red[i];
    const float rinv = rsqrtf(tot / (float)D_ + 1e-6f);
    for (int d = threadIdx.x; d < D_; d += 256)
        out[(size_t)t * D_ + d] = (_Float16)(xr[d] * rinv * wr[d]);
}

// ---------------- RoPE + layout to head-major f16 --------------------------
__global__ __launch_bounds__(256) void rope_kernel(
    const float* __restrict__ qkv, const int* __restrict__ pos_ids,
    _Float16* __restrict__ qf, _Float16* __restrict__ kf,
    _Float16* __restrict__ vf) {
    const int t = blockIdx.x;
    const int b = t / S_;
    const int s = t - b * S_;
    const float pos = (float)pos_ids[(size_t)b * S_ + s];
    const float* row = qkv + (size_t)t * QKVN;
    const float LOG_THETA = 13.815510558f;   // ln(1e6)

    for (int idx = threadIdx.x; idx < QKVN; idx += 256) {
        if (idx < HQ_ * HD_ + HKV_ * HD_) {        // q or k : rotate
            const int base = (idx / HD_) * HD_;
            const int d    = idx & (HD_ - 1);
            const int dm   = d & 31;
            const float inv = __expf(-(float)dm * (LOG_THETA / 32.0f));
            const float f   = pos * inv;
            float sn, cs;
            __sincosf(f, &sn, &cs);
            const float v  = row[idx];
            const float pt = (d < 32) ? -row[base + d + 32] : row[base + d - 32];
            const float o  = v * cs + pt * sn;
            if (idx < HQ_ * HD_) {
                const int h = idx / HD_;
                qf[(((size_t)b * HQ_ + h) * S_ + s) * HD_ + d] = (_Float16)o;
            } else {
                const int h = (idx - HQ_ * HD_) / HD_;
                kf[(((size_t)b * HKV_ + h) * S_ + s) * HD_ + d] = (_Float16)o;
            }
        } else {                                    // v : copy
            const int h = (idx - HQ_ * HD_ - HKV_ * HD_) / HD_;
            const int d = idx & (HD_ - 1);
            vf[(((size_t)b * HKV_ + h) * S_ + s) * HD_ + d] = (_Float16)row[idx];
        }
    }
}

// ---------------- generic WMMA GEMM: C = A[f16] * W[f32]^T + epilogue ------
// Block tile 64(M) x 128(N), K-step 32.  8 waves, each computes 16x64
// (4 WMMAs per K-step against a single A-fragment -> high matrix density).
#define EPI_BIAS 0
#define EPI_RES  1
#define EPI_NONE 2
#define EPI_SILU 3
#define EPI_DOWN 4

template <int EPI>
__global__ __launch_bounds__(256) void gemm_kernel(
    const _Float16* __restrict__ A, int K,           // A: [M,K] f16 row-major
    const float* __restrict__ W0, const float* __restrict__ W1, // [Nw,K] f32
    const float* __restrict__ b0, const float* __restrict__ b1, // bias (EPI_BIAS)
    float* __restrict__ Cf, _Float16* __restrict__ Ch,
    int ldc, int col_off,
    const float* __restrict__ ex0, const float* __restrict__ ex1) {
    const int m0 = blockIdx.x * 64;
    const int n0 = blockIdx.y * 128;
    const int tid = threadIdx.x;
    // every 64-row tile sits inside one batch & one segment (1088 = 17*64)
    const int seg = ((m0 % S_) >= S1_) ? 1 : 0;
    const float* W = seg ? W1 : W0;

    __shared__ _Float16 sA[64 * 32];    // [row][k]
    __shared__ _Float16 sB[128 * 32];   // [n][k]

    const int wid  = tid >> 5;
    const int wr16 = (wid >> 1) * 16;   // wave row base (4 row-waves)
    const int wc64 = (wid & 1) * 64;    // wave col base (2 col-waves x 64)
    const v8f z = {0.f, 0.f, 0.f, 0.f, 0.f, 0.f, 0.f, 0.f};
    v8f acc[4] = {z, z, z, z};

    const int arow = tid >> 2;          // 0..63   (A staging)
    const int achk = (tid & 3) * 8;     // k-chunk of 8 halves
    const int brow = tid >> 1;          // 0..127  (B staging)
    const int bchk = (tid & 1) * 16;    // k-chunk of 16 floats

    for (int kk = 0; kk < K; kk += 32) {
        __syncthreads();
        // stage A (f16, 16B vector copy)
        {
            const _Float16* ar = A + (size_t)(m0 + arow) * K + kk + achk;
            *(float4*)&sA[arow * 32 + achk] = *(const float4*)ar;
            if (kk + 32 < K) __builtin_prefetch(ar + 32, 0, 0);
        }
        // stage W row slice, converting f32 -> f16 (16 floats / thread)
        {
            const float* wr = W + (size_t)(n0 + brow) * K + kk + bchk;
            _Float16* db = &sB[brow * 32 + bchk];
#pragma unroll
            for (int q = 0; q < 4; ++q) {
                const float4 wv = ((const float4*)wr)[q];
                db[4 * q + 0] = (_Float16)wv.x;
                db[4 * q + 1] = (_Float16)wv.y;
                db[4 * q + 2] = (_Float16)wv.z;
                db[4 * q + 3] = (_Float16)wv.w;
            }
            if (kk + 32 < K) __builtin_prefetch(wr + 32, 0, 0);
        }
        __syncthreads();

        const v16h a = ldfrag(&sA[wr16 * 32], 32);
#pragma unroll
        for (int sub = 0; sub < 4; ++sub) {
            const v16h bf = ldfrag(&sB[(wc64 + sub * 16) * 32], 32);
            acc[sub] = wmma_f16(a, bf, acc[sub]);
        }
    }

    // epilogue / store.  C layout: lane = N (l%16), VGPR r = row r (+8 hi half)
    const int l     = tid & 31;
    const int half8 = (l & 16) ? 8 : 0;
    const int nl    = l & 15;
#pragma unroll
    for (int sub = 0; sub < 4; ++sub) {
#pragma unroll
        for (int r = 0; r < 8; ++r) {
            const int t = m0 + wr16 + r + half8;        // token row
            const int n = n0 + wc64 + sub * 16 + nl;    // weight-local column
            float c = acc[sub][r];
            if (EPI == EPI_BIAS) {
                c += (seg ? b1 : b0)[n];
                Cf[(size_t)t * ldc + col_off + n] = c;
            } else if (EPI == EPI_RES) {
                const int b = t / S_, s = t - (t / S_) * S_;
                const float* xr = (s < S1_)
                    ? ex0 + (size_t)(b * S1_ + s) * D_
                    : ex1 + (size_t)(b * S2_ + (s - S1_)) * D_;
                Cf[(size_t)t * ldc + n] = c + xr[n];
            } else if (EPI == EPI_NONE) {
                Cf[(size_t)t * ldc + col_off + n] = c;
            } else if (EPI == EPI_SILU) {
                const float g  = ex0[(size_t)t * FF_ + n];
                const float sg = g / (1.0f + __expf(-g));
                Ch[(size_t)t * FF_ + n] = (_Float16)(sg * c);
            } else { // EPI_DOWN: += residual, remap to (out0, out1) tuple
                const int b = t / S_, s = t - (t / S_) * S_;
                const size_t off = (s < S1_)
                    ? (size_t)(b * S1_ + s) * D_ + n
                    : (size_t)B_ * S1_ * D_ + (size_t)(b * S2_ + (s - S1_)) * D_ + n;
                Cf[off] = c + ex0[(size_t)t * D_ + n];
            }
        }
    }
}

// ---------------- flash attention: 1 wave per (b, h, 16-query tile) --------
__global__ __launch_bounds__(32) void attn_kernel(
    const _Float16* __restrict__ qf, const _Float16* __restrict__ kf,
    const _Float16* __restrict__ vf, _Float16* __restrict__ att) {
    const int id  = blockIdx.x;
    const int qt  = id % (S_ / 16);
    const int h   = (id / (S_ / 16)) % HQ_;
    const int b   = id / ((S_ / 16) * HQ_);
    const int kvh = h / (HQ_ / HKV_);
    const int qs  = qt * 16;
    const int l   = threadIdx.x;
    const int half8 = (l & 16) ? 8 : 0;
    const int nl    = l & 15;

    __shared__ _Float16 k_lds[32][64];   // [key][d]
    __shared__ _Float16 v_lds[64][32];   // [d][key]  (transposed for PV B-frag)
    __shared__ _Float16 p_lds[16][32];   // P restage C-layout -> A-layout

    const _Float16* qbase = qf + (((size_t)b * HQ_ + h) * S_ + qs) * HD_;
    const v16h aq0 = ldfrag(qbase,      HD_);   // K-dim 0..31
    const v16h aq1 = ldfrag(qbase + 32, HD_);   // K-dim 32..63

    const _Float16* kbase = kf + (((size_t)b * HKV_ + kvh) * S_) * HD_;
    const _Float16* vbase = vf + (((size_t)b * HKV_ + kvh) * S_) * HD_;

    const v8f z = {0.f, 0.f, 0.f, 0.f, 0.f, 0.f, 0.f, 0.f};
    v8f accO[4] = {z, z, z, z};
    float m_arr[8], l_arr[8];
#pragma unroll
    for (int r = 0; r < 8; ++r) { m_arr[r] = NEG_BIG; l_arr[r] = 0.f; }

    for (int kt = 0; kt < S_ / 32; ++kt) {
        const int ks = kt * 32;
        // prefix-LM mask (analytic): prefix q -> prefix keys only;
        // suffix q -> all prefix keys + causal suffix keys.
        if ((ks >= S1_) && (qs < S1_ || ks > qs + 15)) continue;

        // stage K tile (natural layout) and V tile (transposed)
        {
            const _Float16* kr = kbase + (size_t)(ks + l) * HD_;
            float4* kd = (float4*)&k_lds[l][0];
#pragma unroll
            for (int i = 0; i < 8; ++i) kd[i] = ((const float4*)kr)[i];
            const _Float16* vr = vbase + (size_t)(ks + l) * HD_;
#pragma unroll
            for (int d = 0; d < 64; ++d) v_lds[d][l] = vr[d];
        }
        __syncthreads();

        // scores: 16 q x 32 keys, K-dim = 64 -> 4 WMMAs
        v8f s0 = z, s1 = z;
        s0 = wmma_f16(aq0, ldfrag(&k_lds[0][0],       HD_), s0);
        s0 = wmma_f16(aq1, ldfrag(&k_lds[0][0] + 32,  HD_), s0);
        s1 = wmma_f16(aq0, ldfrag(&k_lds[16][0],      HD_), s1);
        s1 = wmma_f16(aq1, ldfrag(&k_lds[16][0] + 32, HD_), s1);

        float sc0[8], sc1[8];
        if (ks < S1_) {
            // prefix keys: never masked -> pure scale path (32 of 34 tiles)
#pragma unroll
            for (int r = 0; r < 8; ++r) {
                sc0[r] = s0[r] * 0.125f;   // HD^-0.5
                sc1[r] = s1[r] * 0.125f;
            }
        } else {
            // suffix keys: causal mask within the tile
#pragma unroll
            for (int r = 0; r < 8; ++r) {
                const int rowg = qs + r + half8;
                const int c0 = ks + nl, c1 = ks + 16 + nl;
                float v0 = s0[r] * 0.125f;
                float v1 = s1[r] * 0.125f;
                if (c0 > rowg) v0 = NEG_BIG;
                if (c1 > rowg) v1 = NEG_BIG;
                sc0[r] = v0; sc1[r] = v1;
            }
        }
        // online softmax per row (rows live in 16-lane halves)
#pragma unroll
        for (int r = 0; r < 8; ++r) {
            const float rm  = fmaxf(redmax16(sc0[r]), redmax16(sc1[r]));
            const float mn  = fmaxf(m_arr[r], rm);
            const float scl = __expf(m_arr[r] - mn);
            const float p0  = __expf(sc0[r] - mn);
            const float p1  = __expf(sc1[r] - mn);
            l_arr[r] = l_arr[r] * scl + redsum16(p0) + redsum16(p1);
            m_arr[r] = mn;
#pragma unroll
            for (int nn = 0; nn < 4; ++nn) accO[nn][r] *= scl;
            p_lds[r + half8][nl]      = (_Float16)p0;
            p_lds[r + half8][16 + nl] = (_Float16)p1;
        }
        __syncthreads();

        // O += P(16x32) x V(32x64): 4 WMMAs
        const v16h ap = ldfrag(&p_lds[0][0], 32);
#pragma unroll
        for (int nn = 0; nn < 4; ++nn)
            accO[nn] = wmma_f16(ap, ldfrag(&v_lds[nn * 16][0], 32), accO[nn]);
        __syncthreads();
    }

    // write O / l  to att[b][s][h*64 + d]  (f16, token-major for O-proj GEMM)
#pragma unroll
    for (int nn = 0; nn < 4; ++nn)
#pragma unroll
        for (int r = 0; r < 8; ++r) {
            const int rowg = qs + r + half8;
            const float val = accO[nn][r] / l_arr[r];
            att[((size_t)b * S_ + rowg) * (HQ_ * HD_) + h * HD_ + nn * 16 + nl] =
                (_Float16)val;
        }
}

// ---------------------------------------------------------------------------
extern "C" void kernel_launch(void* const* d_in, const int* in_sizes, int n_in,
                              void* d_out, int out_size, void* d_ws, size_t ws_size,
                              hipStream_t stream) {
    const float* x0     = (const float*)d_in[0];
    const float* x1     = (const float*)d_in[1];
    const float* ln_w   = (const float*)d_in[2];
    const float* q_w    = (const float*)d_in[3];
    const float* q_b    = (const float*)d_in[4];
    const float* k_w    = (const float*)d_in[5];
    const float* k_b    = (const float*)d_in[6];
    const float* v_w    = (const float*)d_in[7];
    const float* v_b    = (const float*)d_in[8];
    const float* o_w    = (const float*)d_in[9];
    const float* pln_w  = (const float*)d_in[10];
    const float* gate_w = (const float*)d_in[11];
    const float* up_w   = (const float*)d_in[12];
    const float* down_w = (const float*)d_in[13];
    const int*   pos_id = (const int*)d_in[14];
    (void)in_sizes; (void)n_in; (void)out_size; (void)ws_size;

    // workspace carve-out
    char* p = (char*)d_ws;
    auto alloc = [&](size_t bytes) {
        void* r = (void*)p;
        p += (bytes + 255) & ~(size_t)255;
        return r;
    };
    _Float16* h1   = (_Float16*)alloc((size_t)NTOK * D_ * 2);
    float*    qkv  = (float*)   alloc((size_t)NTOK * QKVN * 4);
    _Float16* qf   = (_Float16*)alloc((size_t)NTOK * HQ_ * HD_ * 2);
    _Float16* kf   = (_Float16*)alloc((size_t)NTOK * HKV_ * HD_ * 2);
    _Float16* vf   = (_Float16*)alloc((size_t)NTOK * HKV_ * HD_ * 2);
    _Float16* att  = (_Float16*)alloc((size_t)NTOK * HQ_ * HD_ * 2);
    float*    res  = (float*)   alloc((size_t)NTOK * D_ * 4);
    _Float16* h2   = (_Float16*)alloc((size_t)NTOK * D_ * 2);
    float*    gbuf = (float*)   alloc((size_t)NTOK * FF_ * 4);
    _Float16* aFF  = (_Float16*)alloc((size_t)NTOK * FF_ * 2);

    const int MT = NTOK / 64;   // 68 M-tiles

    // 1) pre-attention RMSNorm -> f16
    rmsnorm_kernel<<<NTOK, 256, 0, stream>>>(x0, x1, ln_w, h1);

    // 2) QKV projections (+bias), columns packed [q | k | v] into qkv f32
    gemm_kernel<EPI_BIAS><<<dim3(MT, HQ_ * HD_ / 128), 256, 0, stream>>>(
        h1, D_, q_w, q_w + (size_t)HQ_ * HD_ * D_, q_b, q_b + HQ_ * HD_,
        qkv, nullptr, QKVN, 0, nullptr, nullptr);
    gemm_kernel<EPI_BIAS><<<dim3(MT, 2 * HKV_ * HD_ / 128 / 2), 256, 0, stream>>>(
        h1, D_, k_w, k_w + (size_t)HKV_ * HD_ * D_, k_b, k_b + HKV_ * HD_,
        qkv, nullptr, QKVN, HQ_ * HD_, nullptr, nullptr);
    gemm_kernel<EPI_BIAS><<<dim3(MT, HKV_ * HD_ / 128), 256, 0, stream>>>(
        h1, D_, v_w, v_w + (size_t)HKV_ * HD_ * D_, v_b, v_b + HKV_ * HD_,
        qkv, nullptr, QKVN, HQ_ * HD_ + HKV_ * HD_, nullptr, nullptr);

    // 3) RoPE + head-major f16 layout
    rope_kernel<<<NTOK, 256, 0, stream>>>(qkv, pos_id, qf, kf, vf);

    // 4) flash attention (prefix-LM mask), f16 output token-major
    attn_kernel<<<B_ * HQ_ * (S_ / 16), 32, 0, stream>>>(qf, kf, vf, att);

    // 5) O projection + residual(x) -> res f32
    gemm_kernel<EPI_RES><<<dim3(MT, D_ / 128), 256, 0, stream>>>(
        att, HQ_ * HD_, o_w, o_w + (size_t)D_ * HQ_ * HD_, nullptr, nullptr,
        res, nullptr, D_, 0, x0, x1);

    // 6) post-attention RMSNorm -> f16
    rmsnorm_kernel<<<NTOK, 256, 0, stream>>>(res, nullptr, pln_w, h2);

    // 7) gate GEMM -> f32
    gemm_kernel<EPI_NONE><<<dim3(MT, FF_ / 128), 256, 0, stream>>>(
        h2, D_, gate_w, gate_w + (size_t)FF_ * D_, nullptr, nullptr,
        gbuf, nullptr, FF_, 0, nullptr, nullptr);

    // 8) up GEMM fused with silu(gate)*up -> f16
    gemm_kernel<EPI_SILU><<<dim3(MT, FF_ / 128), 256, 0, stream>>>(
        h2, D_, up_w, up_w + (size_t)FF_ * D_, nullptr, nullptr,
        nullptr, aFF, FF_, 0, gbuf, nullptr);

    // 9) down GEMM + residual(res), remapped into the (out0, out1) tuple
    gemm_kernel<EPI_DOWN><<<dim3(MT, D_ / 128), 256, 0, stream>>>(
        aFF, FF_, down_w, down_w + (size_t)D_ * FF_, nullptr, nullptr,
        (float*)d_out, nullptr, D_, 0, res, nullptr);
}
